// AttentionLayer_91216515432859
// MI455X (gfx1250) — matrix operand order repaired and checked by reading, not compiled, not run on hardware
//
#include <hip/hip_runtime.h>
#include <hip/hip_bf16.h>
#include <math.h>

typedef __attribute__((ext_vector_type(16))) _Float16 v16h;
typedef __attribute__((ext_vector_type(8)))  _Float16 v8h;
typedef __attribute__((ext_vector_type(8)))  float    v8f;

#define TMAX 224   // padded T (multiple of 32, >= T)
#define DD   64
#define HH1  80
#define HH2  40
#define K1   128   // reduced inner dim for GEMM1 (q / q-k terms folded into c1)
#define CH   32    // t-rows per chunk

// LDS row strides (halves / floats), padded for bank-conflict-free ds_load_b128
#define IST  136   // sInfo  [32][IST]  f16
#define W1ST 136   // sW1T   [80][W1ST] f16
#define H1ST 104   // sH1    [32][H1ST] f16  (K padded to 96)
#define W2ST 104   // sW2T   [48][W2ST] f16
#define H2ST 49    // sH2    [32][H2ST] f32
#define AST  40    // sA1T/sA2T row stride ([n][m] transposed alpha, 20-bank step)

__launch_bounds__(256, 1)
__global__ void din_attention_kernel(
    const float* __restrict__ Q,   // [B, 64]
    const float* __restrict__ Kk,  // [B, T, 64]
    const float* __restrict__ Vv,  // [B, T, 64]
    const float* __restrict__ W1,  // [256, 80]
    const float* __restrict__ B1,  // [80]
    const float* __restrict__ A1,  // [T, 80]
    const float* __restrict__ W2,  // [80, 40]
    const float* __restrict__ B2,  // [40]
    const float* __restrict__ A2,  // [T, 40]
    const float* __restrict__ Wf,  // [40, 1]
    const float* __restrict__ BF,  // [1]
    float* __restrict__ Out,       // [B, 64]
    int T)
{
    __shared__ __align__(16) _Float16 sW1T[HH1 * W1ST];
    __shared__ __align__(16) _Float16 sW2T[48 * W2ST];
    __shared__ __align__(16) _Float16 sInfo[CH * IST];
    __shared__ __align__(16) _Float16 sH1[CH * H1ST];
    __shared__ __align__(16) _Float16 sA1T[HH1 * AST];  // alpha1 transposed [n][m]
    __shared__ __align__(16) _Float16 sA2T[HH2 * AST];  // alpha2 transposed [n][m]
    __shared__ __align__(16) float sH2[CH * H2ST];      // reused as reduction scratch
    __shared__ float sQ[DD];
    __shared__ float sC1[HH1];
    __shared__ float sB2[HH2];
    __shared__ float sWfS[HH2];
    __shared__ float sBF;
    __shared__ float sLogit[TMAX];

    const int tid  = threadIdx.x;
    const int b    = blockIdx.x;
    const int wave = tid >> 5;
    const int lane = tid & 31;
    const int ln   = lane & 15;
    const int hi   = lane >> 4;   // 0 for lanes 0-15, 1 for lanes 16-31

    // ---------------- Setup (weights folded into LDS, f16) ----------------
    if (tid < DD) sQ[tid] = Q[(size_t)b * DD + tid];

    // Effective W1 for reduced-K GEMM1:
    //   h_pre = c1 + k @ (W1_k - W1_qmk) + (q*k) @ W1_qk
    // stored transposed [n][k] to match the WMMA B-fragment layout.
    for (int idx = tid; idx < HH1 * K1; idx += 256) {
        int n = idx >> 7, kk = idx & 127;
        float v;
        if (kk < 64) v = W1[(64 + kk) * HH1 + n] - W1[(128 + kk) * HH1 + n];
        else         v = W1[(192 + (kk - 64)) * HH1 + n];
        sW1T[n * W1ST + kk] = (_Float16)v;
    }
    for (int idx = tid; idx < 48 * 96; idx += 256) {
        int n = idx / 96, kk = idx % 96;
        float v = (n < HH2 && kk < HH1) ? W2[kk * HH2 + n] : 0.f;
        sW2T[n * W2ST + kk] = (_Float16)v;
    }
    if (tid < HH2) { sWfS[tid] = Wf[tid]; sB2[tid] = B2[tid]; }
    if (tid == 0)  sBF = BF[0];
    // zero static K-pad columns 80..95 of h1 (never overwritten)
    for (int idx = tid; idx < CH * 16; idx += 256)
        sH1[(idx >> 4) * H1ST + 80 + (idx & 15)] = (_Float16)0.f;
    __syncthreads();

    // c1[n] = b1[n] + sum_i q[i] * (W1a[i][n] + W1c[i][n])
    if (tid < HH1) {
        float s = B1[tid];
        for (int i = 0; i < 64; ++i)
            s += sQ[i] * (W1[i * HH1 + tid] + W1[(128 + i) * HH1 + tid]);
        sC1[tid] = s;
    }
    __syncthreads();

    // ---------------- Chunked MLP over the T axis ----------------
    const int nChunks = (T + CH - 1) / CH;
    const int tClamp  = T - 1;
    for (int c = 0; c < nChunks; ++c) {
        const int t0 = c * CH;

        // build info' = [k | q*k] tile (f16); padded rows read clamped data
        // (their logits are masked to -inf later, rows are independent)
        for (int idx = tid; idx < CH * DD; idx += 256) {
            int m = idx >> 6, d = idx & 63;
            int t = t0 + m; t = t > tClamp ? tClamp : t;
            float kv = Kk[((size_t)b * T + t) * DD + d];
            sInfo[m * IST + d]      = (_Float16)kv;
            sInfo[m * IST + 64 + d] = (_Float16)(kv * sQ[d]);
        }
        // stage PReLU alphas transposed [n][m]: coalesced global reads,
        // conflict-free vector reads in the WMMA epilogues
        for (int idx = tid; idx < CH * HH1; idx += 256) {
            int m = idx / HH1, n = idx % HH1;
            int t = t0 + m; t = t > tClamp ? tClamp : t;
            sA1T[n * AST + m] = (_Float16)A1[(size_t)t * HH1 + n];
        }
        for (int idx = tid; idx < CH * HH2; idx += 256) {
            int m = idx / HH2, n = idx % HH2;
            int t = t0 + m; t = t > tClamp ? tClamp : t;
            sA2T[n * AST + m] = (_Float16)A2[(size_t)t * HH2 + n];
        }
        // prefetch this chunk's v rows for the tail weighted-sum
        if (tid < 64) {
            int t = t0 + (tid >> 1); t = t > tClamp ? tClamp : t;
            __builtin_prefetch(&Vv[((size_t)b * T + t) * DD + (tid & 1) * 32], 0, 1);
        }
        __syncthreads();

        // GEMM1: [32 x 128] x [128 x 80] -> PReLU -> sH1 (f16)
        for (int tile = wave; tile < 10; tile += 8) {
            int mt = tile & 1, nt = tile >> 1;
            v8f acc = {};
            #pragma unroll
            for (int kk = 0; kk < 4; ++kk) {
                int arow = mt * 16 + ln;
                const v8h a0 = *(const v8h*)&sInfo[arow * IST + kk * 32 + hi * 8];
                const v8h a1 = *(const v8h*)&sInfo[arow * IST + kk * 32 + hi * 8 + 16];
                int brow = nt * 16 + ln;
                const v8h b0 = *(const v8h*)&sW1T[brow * W1ST + kk * 32 + hi * 8];
                const v8h b1 = *(const v8h*)&sW1T[brow * W1ST + kk * 32 + hi * 8 + 16];
                v16h av, bv;
                #pragma unroll
                for (int i = 0; i < 8; ++i) {
                    av[i] = a0[i]; av[i + 8] = a1[i];
                    bv[i] = b0[i]; bv[i + 8] = b1[i];
                }
                acc = __builtin_amdgcn_wmma_f32_16x16x32_f16(
                    false, av, false, bv, (short)0, acc, false, false);
            }
            int n = nt * 16 + ln;
            float cn = sC1[n];
            const v8h alv = *(const v8h*)&sA1T[n * AST + mt * 16 + hi * 8];
            #pragma unroll
            for (int r = 0; r < 8; ++r) {
                int m = mt * 16 + hi * 8 + r;
                float x = acc[r] + cn;
                float al = (float)alv[r];
                x = x > 0.f ? x : al * x;
                sH1[m * H1ST + n] = (_Float16)x;
            }
        }
        __syncthreads();

        // GEMM2: [32 x 96] x [96 x 48] -> PReLU -> sH2 (f32)
        for (int tile = wave; tile < 6; tile += 8) {
            int mt = tile & 1, nt = tile >> 1;
            v8f acc = {};
            #pragma unroll
            for (int kk = 0; kk < 3; ++kk) {
                int arow = mt * 16 + ln;
                const v8h a0 = *(const v8h*)&sH1[arow * H1ST + kk * 32 + hi * 8];
                const v8h a1 = *(const v8h*)&sH1[arow * H1ST + kk * 32 + hi * 8 + 16];
                int brow = nt * 16 + ln;
                const v8h b0 = *(const v8h*)&sW2T[brow * W2ST + kk * 32 + hi * 8];
                const v8h b1 = *(const v8h*)&sW2T[brow * W2ST + kk * 32 + hi * 8 + 16];
                v16h av, bv;
                #pragma unroll
                for (int i = 0; i < 8; ++i) {
                    av[i] = a0[i]; av[i + 8] = a1[i];
                    bv[i] = b0[i]; bv[i + 8] = b1[i];
                }
                acc = __builtin_amdgcn_wmma_f32_16x16x32_f16(
                    false, av, false, bv, (short)0, acc, false, false);
            }
            int n = nt * 16 + ln;
            int nc = n < HH2 ? n : HH2 - 1;          // pad columns: harmless values
            float b2v = sB2[nc];
            const v8h alv = *(const v8h*)&sA2T[nc * AST + mt * 16 + hi * 8];
            #pragma unroll
            for (int r = 0; r < 8; ++r) {
                int m = mt * 16 + hi * 8 + r;
                float x = acc[r] + b2v;
                float al = (float)alv[r];
                x = x > 0.f ? x : al * x;
                sH2[m * H2ST + n] = x;
            }
        }
        __syncthreads();

        // final Dense(1): logits[t] = h2[t] . Wf + bf
        if (tid < CH) {
            int t = t0 + tid;
            float s = sBF;
            for (int n = 0; n < HH2; ++n)
                s += sH2[tid * H2ST + n] * sWfS[n];
            if (t < TMAX) sLogit[t] = s;
        }
        __syncthreads();
    }

    // mask padded rows
    if (tid >= T && tid < TMAX) sLogit[tid] = -3.0e38f;
    __syncthreads();

    // ---------------- Softmax over T (sH2 reused as scratch) ----------------
    sH2[tid] = (tid < TMAX) ? sLogit[tid] : -3.0e38f;
    __syncthreads();
    for (int s = 128; s > 0; s >>= 1) {
        if (tid < s) sH2[tid] = fmaxf(sH2[tid], sH2[tid + s]);
        __syncthreads();
    }
    float mx = sH2[0];
    __syncthreads();
    float e = (tid < T) ? __expf(sLogit[tid] - mx) : 0.f;
    sH2[tid] = e;
    __syncthreads();
    for (int s = 128; s > 0; s >>= 1) {
        if (tid < s) sH2[tid] += sH2[tid + s];
        __syncthreads();
    }
    float rinv = 1.f / sH2[0];
    if (tid < TMAX) sLogit[tid] = e * rinv;
    __syncthreads();

    // ---------------- out[b] = sum_t w[t] * v[b,t,:] ----------------
    {
        int g = tid >> 6, d = tid & 63;
        float accv = 0.f;
        for (int t = g; t < T; t += 4)
            accv += sLogit[t] * Vv[((size_t)b * T + t) * DD + d];
        sH2[g * DD + d] = accv;   // partials in scratch
    }
    __syncthreads();
    if (tid < DD)
        Out[(size_t)b * DD + tid] =
            sH2[tid] + sH2[DD + tid] + sH2[2 * DD + tid] + sH2[3 * DD + tid];
}

extern "C" void kernel_launch(void* const* d_in, const int* in_sizes, int n_in,
                              void* d_out, int out_size, void* d_ws, size_t ws_size,
                              hipStream_t stream) {
    (void)n_in; (void)out_size; (void)d_ws; (void)ws_size;
    const float* Q  = (const float*)d_in[0];
    const float* Kk = (const float*)d_in[1];
    const float* Vv = (const float*)d_in[2];
    const float* W1 = (const float*)d_in[3];
    const float* B1 = (const float*)d_in[4];
    const float* A1 = (const float*)d_in[5];
    const float* W2 = (const float*)d_in[6];
    const float* B2 = (const float*)d_in[7];
    const float* A2 = (const float*)d_in[8];
    const float* Wf = (const float*)d_in[9];
    const float* BF = (const float*)d_in[10];
    float* Out = (float*)d_out;

    const int B = in_sizes[0] / DD;                 // 2048
    const int T = in_sizes[1] / in_sizes[0];        // 200

    din_attention_kernel<<<B, 256, 0, stream>>>(
        Q, Kk, Vv, W1, B1, A1, W2, B2, A2, Wf, BF, Out, T);
}
